// ResidualAttentionLayer_30039001268721
// MI455X (gfx1250) — matrix operand order repaired and checked by reading, not compiled, not run on hardware
//
#include <hip/hip_runtime.h>
#include <hip/hip_bf16.h>
#include <math.h>

// ---------------- constants ----------------
#define BB   4
#define TT   1024
#define DD   1024
#define HH   16
#define HD   64
#define FF   4096
#define BT   (BB*TT)        // 4096 rows
#define SCALE 0.125f        // HD^-0.5

typedef __attribute__((ext_vector_type(16))) _Float16 v16h;
typedef __attribute__((ext_vector_type(8)))  _Float16 v8h;
typedef __attribute__((ext_vector_type(8)))  float    v8f;

// ---------------- WMMA helpers ----------------
__device__ inline v8f wmma32(v16h a, v16h b, v8f c) {
  // D = A(16x32 f16) * B(32x16 f16) + C(16x16 f32)
  return __builtin_amdgcn_wmma_f32_16x16x32_f16(false, a, false, b, (short)0, c,
                                                false, false);
}

// A-matrix fragment (16x32, lane = M row):
//  lanes 0-15:  halfs 0-7 = K0..7,  halfs 8-15 = K16..23
//  lanes 16-31: halfs 0-7 = K8..15, halfs 8-15 = K24..31
// rowp points at K=0 of this lane's row; hf = lane>>4.
__device__ inline v16h frag_a(const _Float16* rowp, int hf) {
  v8h lo = *(const v8h*)(rowp + hf * 8);
  v8h hi = *(const v8h*)(rowp + 16 + hf * 8);
  return __builtin_shufflevector(lo, hi, 0,1,2,3,4,5,6,7,8,9,10,11,12,13,14,15);
}

// B-matrix fragment (32x16, lane = N col):
//  lanes 0-15:  halfs 0-7 = K0..7,  halfs 8-15 = K16..23 ; lanes 16-31: K8..15, K24..31
// rowp points at K=0 of this lane's column (stored as row of B^T).
__device__ inline v16h frag_b(const _Float16* rowp, int hf) {
  v8h lo = *(const v8h*)(rowp + hf * 8);
  v8h hi = *(const v8h*)(rowp + 16 + hf * 8);
  return __builtin_shufflevector(lo, hi, 0,1,2,3,4,5,6,7,8,9,10,11,12,13,14,15);
}

// ---------------- weight transpose + f32->f16 convert ----------------
// in: f32 [R][C] row-major.  out: f16 [C][R] row-major (i.e. B^T with K contiguous).
__global__ __launch_bounds__(256) void transpose_k(const float* __restrict__ in,
                                                   _Float16* __restrict__ out,
                                                   int R, int C) {
  __shared__ float tile[32][33];
  int c0 = blockIdx.x * 32, r0 = blockIdx.y * 32;
  for (int i = threadIdx.y; i < 32; i += 8)
    tile[i][threadIdx.x] = in[(size_t)(r0 + i) * C + c0 + threadIdx.x];
  __syncthreads();
  for (int i = threadIdx.y; i < 32; i += 8)
    out[(size_t)(c0 + i) * R + r0 + threadIdx.x] = (_Float16)tile[threadIdx.x][i];
}

// ---------------- layernorm (f32 in -> f16 out), one row per block ----------------
__global__ __launch_bounds__(256) void layernorm_k(const float* __restrict__ x,
                                                   const float* __restrict__ g,
                                                   const float* __restrict__ be,
                                                   _Float16* __restrict__ out) {
  __shared__ float red[256];
  int row = blockIdx.x, tid = threadIdx.x;
  const float* xr = x + (size_t)row * DD;
  float v0[4];
  float s = 0.f;
  for (int i = 0; i < 4; ++i) { v0[i] = xr[tid + i * 256]; s += v0[i]; }
  red[tid] = s; __syncthreads();
  for (int st = 128; st > 0; st >>= 1) { if (tid < st) red[tid] += red[tid + st]; __syncthreads(); }
  float mu = red[0] * (1.f / DD);
  __syncthreads();
  float vs = 0.f;
  for (int i = 0; i < 4; ++i) { float d = v0[i] - mu; vs += d * d; }
  red[tid] = vs; __syncthreads();
  for (int st = 128; st > 0; st >>= 1) { if (tid < st) red[tid] += red[tid + st]; __syncthreads(); }
  float rstd = rsqrtf(red[0] * (1.f / DD) + 1e-5f);
  for (int i = 0; i < 4; ++i) {
    int c = tid + i * 256;
    out[(size_t)row * DD + c] = (_Float16)((v0[i] - mu) * rstd * g[c] + be[c]);
  }
}

// ---------------- generic WMMA GEMM:  C = A(f16 MxK) * Bt(f16 NxK)^T ----------------
// block tile 64x256, 8 waves (2x4), wave tile 32x64 (2x4 WMMA of 16x16), K-step 32.
struct GemmP {
  const _Float16* A;
  const _Float16* Bt;
  int M, N, K;
  const float* bias;
  const float* resid;
  float*     outf;
  _Float16*  outh;
  _Float16 *q, *k, *vt;
};

#define LDS_STRIDE 40  // 80B rows: 16B aligned, bank-conflict friendly

template<int EPI>
__global__ __launch_bounds__(256) void gemm_f16(GemmP p) {
  __shared__ _Float16 As[64 * LDS_STRIDE];
  __shared__ _Float16 Bs[256 * LDS_STRIDE];
  int tid  = threadIdx.x;
  int lane = tid & 31, wid = tid >> 5;
  int n15  = lane & 15, hf = lane >> 4;
  int wm   = wid >> 2,  wn = wid & 3;
  int m0b  = blockIdx.y * 64, n0b = blockIdx.x * 256;

  v8f acc[2][4] = {};
  int arow = tid >> 2, acol = (tid & 3) * 8;

  for (int kb = 0; kb < p.K; kb += 32) {
    // stage A tile (64x32) and Bt tile (256x32) into LDS, 16B per load
    *(v8h*)&As[arow * LDS_STRIDE + acol] =
        *(const v8h*)(p.A + (size_t)(m0b + arow) * p.K + kb + acol);
#pragma unroll
    for (int i = 0; i < 4; ++i) {
      int brow = arow + i * 64;
      *(v8h*)&Bs[brow * LDS_STRIDE + acol] =
          *(const v8h*)(p.Bt + (size_t)(n0b + brow) * p.K + kb + acol);
    }
    __syncthreads();
    v16h af[2], bf[4];
#pragma unroll
    for (int mt = 0; mt < 2; ++mt)
      af[mt] = frag_a(&As[(wm * 32 + mt * 16 + n15) * LDS_STRIDE], hf);
#pragma unroll
    for (int nt = 0; nt < 4; ++nt)
      bf[nt] = frag_b(&Bs[(wn * 64 + nt * 16 + n15) * LDS_STRIDE], hf);
#pragma unroll
    for (int mt = 0; mt < 2; ++mt)
#pragma unroll
      for (int nt = 0; nt < 4; ++nt)
        acc[mt][nt] = wmma32(af[mt], bf[nt], acc[mt][nt]);
    __syncthreads();
  }

  // epilogue: D layout -> lane (0-15): N=n15, M=r ; lanes 16-31: M=8+r
#pragma unroll
  for (int mt = 0; mt < 2; ++mt)
#pragma unroll
    for (int nt = 0; nt < 4; ++nt)
#pragma unroll
      for (int r = 0; r < 8; ++r) {
        int m = m0b + wm * 32 + mt * 16 + hf * 8 + r;
        int n = n0b + wn * 64 + nt * 16 + n15;
        float v = acc[mt][nt][r];
        if (EPI == 0) {
          // QKV split: n in [0,3D). q,k -> [B,H,T,HD]; v -> transposed [B,H,HD,T]
          int which = n >> 10, d = n & 1023;
          int h = d >> 6, hd = d & 63;
          int b = m >> 10, t = m & 1023;
          _Float16 hv = (_Float16)v;
          if (which == 0)      p.q [((size_t)(b * HH + h) * TT + t) * HD + hd] = hv;
          else if (which == 1) p.k [((size_t)(b * HH + h) * TT + t) * HD + hd] = hv;
          else                 p.vt[((size_t)(b * HH + h) * HD + hd) * TT + t] = hv;
        } else if (EPI == 1) {
          // out-projection + residual -> f32
          p.outf[(size_t)m * p.N + n] = v + p.resid[(size_t)m * p.N + n];
        } else if (EPI == 2) {
          // bias + exact GELU -> f16
          float t = v + p.bias[n];
          p.outh[(size_t)m * p.N + n] =
              (_Float16)(0.5f * t * (1.f + erff(t * 0.70710678118f)));
        } else {
          // final: bias + residual -> f32 (d_out)
          p.outf[(size_t)m * p.N + n] = v + p.bias[n] + p.resid[(size_t)m * p.N + n];
        }
      }
}

// ---------------- fused attention (flash-style, online softmax) ----------------
// grid = B*H*(T/16) blocks, 1 wave each. Each wave: 16 query rows x all 1024 keys.
// scores = q@k^T*scale + prev_attn is streamed to d_out; P never hits HBM.
__global__ __launch_bounds__(32) void attn_fused(const _Float16* __restrict__ q,
                                                 const _Float16* __restrict__ k,
                                                 const _Float16* __restrict__ vt,
                                                 const float* __restrict__ prev,
                                                 float* __restrict__ scores,
                                                 _Float16* __restrict__ aout) {
  __shared__ _Float16 Plds[16 * LDS_STRIDE];
  int lane = threadIdx.x & 31;
  int n15 = lane & 15, hf = lane >> 4;
  int bid = blockIdx.x;
  int qt = bid & 63, h = (bid >> 6) & 15, b = bid >> 10;
  int row0 = qt * 16;

  const _Float16* Q  = q  + (((size_t)(b * HH + h)) * TT + row0) * HD;
  const _Float16* Kp = k  + ((size_t)(b * HH + h)) * TT * HD;
  const _Float16* Vt = vt + ((size_t)(b * HH + h)) * HD * TT;
  const float*    PA = prev   + (((size_t)(b * HH + h)) * TT + row0) * TT;
  float*          SC = scores + (((size_t)(b * HH + h)) * TT + row0) * TT;

  // Q fragments for K-dim chunks [0,32) and [32,64)
  v16h qf0 = frag_a(Q + n15 * HD, hf);
  v16h qf1 = frag_a(Q + n15 * HD + 32, hf);

  v8f O[4] = {};
  float mrow[8], lrow[8];
#pragma unroll
  for (int r = 0; r < 8; ++r) { mrow[r] = -INFINITY; lrow[r] = 0.f; }

  for (int kc = 0; kc < TT; kc += 32) {
    v8f S[2];
#pragma unroll
    for (int sub = 0; sub < 2; ++sub) {
      int j0 = kc + sub * 16;
      // B = K^T tile: lane = key column, features contiguous
      v16h bf0 = frag_b(Kp + (size_t)(j0 + n15) * HD, hf);
      v16h bf1 = frag_b(Kp + (size_t)(j0 + n15) * HD + 32, hf);
      v8f s = {};
      s = wmma32(qf0, bf0, s);
      s = wmma32(qf1, bf1, s);
#pragma unroll
      for (int r = 0; r < 8; ++r) {
        int mr = hf * 8 + r;
        float val = s[r] * SCALE + PA[(size_t)mr * TT + j0 + n15];
        SC[(size_t)mr * TT + j0 + n15] = val;   // streamed score output
        s[r] = val;
      }
      S[sub] = s;
    }
    // online softmax update across the 32-key chunk
    float corr[8];
#pragma unroll
    for (int r = 0; r < 8; ++r) {
      float mx = fmaxf(S[0][r], S[1][r]);
#pragma unroll
      for (int msk = 1; msk < 16; msk <<= 1)
        mx = fmaxf(mx, __shfl_xor(mx, msk, 32));
      float mn = fmaxf(mrow[r], mx);
      float c  = __expf(mrow[r] - mn);
      float p0 = __expf(S[0][r] - mn);
      float p1 = __expf(S[1][r] - mn);
      float sm = p0 + p1;
#pragma unroll
      for (int msk = 1; msk < 16; msk <<= 1)
        sm += __shfl_xor(sm, msk, 32);
      lrow[r] = lrow[r] * c + sm;
      mrow[r] = mn;
      corr[r] = c;
      int mr = hf * 8 + r;
      Plds[mr * LDS_STRIDE + n15]      = (_Float16)p0;   // D layout -> row-major P
      Plds[mr * LDS_STRIDE + 16 + n15] = (_Float16)p1;
    }
#pragma unroll
    for (int t = 0; t < 4; ++t)
#pragma unroll
      for (int r = 0; r < 8; ++r) O[t][r] *= corr[r];
    __syncthreads();   // single-wave WG: orders LDS bounce (barrier -> s_nop + fence)
    v16h pf = frag_a(&Plds[n15 * LDS_STRIDE], hf);
#pragma unroll
    for (int t = 0; t < 4; ++t) {
      // B = V chunk: lane = output feature col, keys contiguous (V stored transposed)
      v16h vf = frag_b(Vt + (size_t)(t * 16 + n15) * TT + kc, hf);
      O[t] = wmma32(pf, vf, O[t]);
    }
    __syncthreads();
  }
  // normalize and write attn output as [B,T,D] f16 for the out-projection GEMM
#pragma unroll
  for (int t = 0; t < 4; ++t)
#pragma unroll
    for (int r = 0; r < 8; ++r) {
      int mr = hf * 8 + r;
      float val = O[t][r] / lrow[r];
      aout[((size_t)(b * TT) + row0 + mr) * DD + h * HD + t * 16 + n15] = (_Float16)val;
    }
}

// ---------------- host launcher ----------------
extern "C" void kernel_launch(void* const* d_in, const int* in_sizes, int n_in,
                              void* d_out, int out_size, void* d_ws, size_t ws_size,
                              hipStream_t stream) {
  (void)in_sizes; (void)n_in; (void)out_size; (void)ws_size;
  const float* x    = (const float*)d_in[0];
  const float* prev = (const float*)d_in[1];
  const float* Wqkv = (const float*)d_in[2];
  const float* Wout = (const float*)d_in[3];
  const float* W1   = (const float*)d_in[4];
  const float* b1   = (const float*)d_in[5];
  const float* W2   = (const float*)d_in[6];
  const float* b2   = (const float*)d_in[7];
  const float* g1   = (const float*)d_in[8];
  const float* be1  = (const float*)d_in[9];
  const float* g2   = (const float*)d_in[10];
  const float* be2  = (const float*)d_in[11];

  float* y      = (float*)d_out;                       // [B,T,D]
  float* scores = y + (size_t)BT * DD;                 // [B,H,T,T]

  char* wsb = (char*)d_ws;
  size_t off = 0;
  auto take = [&](size_t bytes) -> void* {
    void* p = wsb + off;
    off += (bytes + 255) & ~(size_t)255;
    return p;
  };
  _Float16* xn    = (_Float16*)take((size_t)BT * DD * 2);       // LN1 out
  _Float16* wqkvt = (_Float16*)take((size_t)3 * DD * DD * 2);   // [3D][D]
  _Float16* woutt = (_Float16*)take((size_t)DD * DD * 2);       // [D][D]
  _Float16* w1t   = (_Float16*)take((size_t)FF * DD * 2);       // [FF][D]
  _Float16* w2t   = (_Float16*)take((size_t)DD * FF * 2);       // [D][FF]
  _Float16* qb    = (_Float16*)take((size_t)BT * DD * 2);       // [B,H,T,HD]
  _Float16* kb    = (_Float16*)take((size_t)BT * DD * 2);       // [B,H,T,HD]
  _Float16* vtb   = (_Float16*)take((size_t)BT * DD * 2);       // [B,H,HD,T]
  _Float16* aout  = (_Float16*)take((size_t)BT * DD * 2);       // [B,T,D]
  float*    x2    = (float*)   take((size_t)BT * DD * 4);       // residual 1 out
  _Float16* x2n   = (_Float16*)take((size_t)BT * DD * 2);       // LN2 out
  _Float16* hbuf  = (_Float16*)take((size_t)BT * FF * 2);       // GELU out

  dim3 tb(32, 8);
  transpose_k<<<dim3(3 * DD / 32, DD / 32), tb, 0, stream>>>(Wqkv, wqkvt, DD, 3 * DD);
  transpose_k<<<dim3(DD / 32,     DD / 32), tb, 0, stream>>>(Wout, woutt, DD, DD);
  transpose_k<<<dim3(FF / 32,     DD / 32), tb, 0, stream>>>(W1,   w1t,   DD, FF);
  transpose_k<<<dim3(DD / 32,     FF / 32), tb, 0, stream>>>(W2,   w2t,   FF, DD);

  layernorm_k<<<BT, 256, 0, stream>>>(x, g1, be1, xn);

  GemmP pq{xn, wqkvt, BT, 3 * DD, DD, nullptr, nullptr, nullptr, nullptr, qb, kb, vtb};
  gemm_f16<0><<<dim3(3 * DD / 256, BT / 64), 256, 0, stream>>>(pq);

  attn_fused<<<BB * HH * (TT / 16), 32, 0, stream>>>(qb, kb, vtb, prev, scores, aout);

  GemmP po{aout, woutt, BT, DD, DD, nullptr, x, x2, nullptr, nullptr, nullptr, nullptr};
  gemm_f16<1><<<dim3(DD / 256, BT / 64), 256, 0, stream>>>(po);

  layernorm_k<<<BT, 256, 0, stream>>>(x2, g2, be2, x2n);

  GemmP p1{x2n, w1t, BT, FF, DD, b1, nullptr, nullptr, hbuf, nullptr, nullptr, nullptr};
  gemm_f16<2><<<dim3(FF / 256, BT / 64), 256, 0, stream>>>(p1);

  GemmP p2{hbuf, w2t, BT, DD, FF, b2, x2, y, nullptr, nullptr, nullptr, nullptr};
  gemm_f16<3><<<dim3(DD / 256, BT / 64), 256, 0, stream>>>(p2);
}
